// snnSdpaLlamaAttention_80058190397924
// MI455X (gfx1250) — matrix operand order repaired and checked by reading, not compiled, not run on hardware
//
#include <hip/hip_runtime.h>
#include <cstdint>
#include <cstddef>

// ---------------------------------------------------------------------------
// Types for CDNA5 WMMA (wave32): V_WMMA_F32_16X16X32_BF16
// ---------------------------------------------------------------------------
typedef __attribute__((ext_vector_type(16))) __bf16 v16bf;
typedef __attribute__((ext_vector_type(8)))  float  v8f;

#define WMMA_BF16(a, b, c) \
  __builtin_amdgcn_wmma_f32_16x16x32_bf16(false, (a), false, (b), (short)0, (c), false, false)

union BfFrag { v16bf v; uint4 q[2]; };

__device__ __forceinline__ __bf16 f2bf(float f) {
  union { float f; unsigned u; } in; in.f = f;
  unsigned r = in.u + 0x7FFFu + ((in.u >> 16) & 1u);   // round-to-nearest-even
  union { unsigned short s; __bf16 b; } out; out.s = (unsigned short)(r >> 16);
  return out.b;
}

// A-matrix 16x32 bf16 frag: lane m = lane&15 holds K = g8..g8+7 and g8+16..g8+23
__device__ __forceinline__ v16bf load_a_frag(const __bf16* row, int kk, int lane) {
  const int g8 = (lane & 16) ? 8 : 0;
  const uint4* p = reinterpret_cast<const uint4*>(row + kk + g8);
  BfFrag u; u.q[0] = p[0]; u.q[1] = p[2];
  return u.v;
}

// B-matrix 32x16 bf16 frag from a row of B^T: lane group holds K = g8..g8+7 and g8+16..g8+23
__device__ __forceinline__ v16bf load_b_frag(const __bf16* btrow, int kk, int lane) {
  const int g8 = (lane & 16) ? 8 : 0;
  const uint4* p = reinterpret_cast<const uint4*>(btrow + kk + g8);
  BfFrag u; u.q[0] = p[0]; u.q[1] = p[2];
  return u.v;
}

// Build an A-frag of Y = softmax(X) on the fly from fp32 X row + (max, 1/sum)
__device__ __forceinline__ v16bf y_frag(const float* Xr, int kk, int lane,
                                        float mx, float inv) {
  const int g8 = (lane & 16) ? 8 : 0;
  const float4* p = reinterpret_cast<const float4*>(Xr + kk + g8);
  float xs[16];
  *reinterpret_cast<float4*>(xs + 0)  = p[0];
  *reinterpret_cast<float4*>(xs + 4)  = p[1];
  *reinterpret_cast<float4*>(xs + 8)  = p[4];   // +16 floats
  *reinterpret_cast<float4*>(xs + 12) = p[5];
  v16bf a;
#pragma unroll
  for (int j = 0; j < 16; ++j) a[j] = f2bf(__expf(xs[j] - mx) * inv);
  return a;
}

__device__ __forceinline__ float wred_max(float v) {
#pragma unroll
  for (int off = 16; off; off >>= 1) v = fmaxf(v, __shfl_xor(v, off, 32));
  return v;
}
__device__ __forceinline__ float wred_sum(float v) {
#pragma unroll
  for (int off = 16; off; off >>= 1) v += __shfl_xor(v, off, 32);
  return v;
}

// gfx1250 async global->LDS staging: 64 contiguous bytes per thread.
// Same inst offset is applied to both LDS and global address (ISA 15.18.3),
// which matches our equal 16B sub-chunk strides on both sides.
__device__ __forceinline__ void stage_async64(unsigned lds_byte_addr,
                                              const __bf16* gsrc) {
  asm volatile("global_load_async_to_lds_b128 %0, %1, off"
               :: "v"(lds_byte_addr), "v"(gsrc) : "memory");
  asm volatile("global_load_async_to_lds_b128 %0, %1, off offset:16"
               :: "v"(lds_byte_addr), "v"(gsrc) : "memory");
  asm volatile("global_load_async_to_lds_b128 %0, %1, off offset:32"
               :: "v"(lds_byte_addr), "v"(gsrc) : "memory");
  asm volatile("global_load_async_to_lds_b128 %0, %1, off offset:48"
               :: "v"(lds_byte_addr), "v"(gsrc) : "memory");
}
__device__ __forceinline__ void wait_async0() {
  asm volatile("s_wait_asynccnt 0x0" ::: "memory");
}

// ---------------------------------------------------------------------------
// fp32 -> bf16 convert
// ---------------------------------------------------------------------------
__global__ void cvt_bf16_kernel(const float* __restrict__ src,
                                __bf16* __restrict__ dst, int n) {
  int i = blockIdx.x * blockDim.x + threadIdx.x;
  if (i < n) dst[i] = f2bf(src[i]);
}

// ---------------------------------------------------------------------------
// C(MxN) fp32 = A(MxK) bf16 @ W(NxK)^T bf16.  Block: 128 threads = 4 waves,
// each wave a 16x128 tile.  W k-slab double-buffered in LDS via async loads:
//   iter i: s_wait_asynccnt 0; barrier; issue async for slab i+1; WMMA slab i.
// ---------------------------------------------------------------------------
__global__ void __launch_bounds__(128)
gemm_xwt_kernel(const __bf16* __restrict__ A, const __bf16* __restrict__ W,
                float* __restrict__ C, int M, int N, int K) {
  __shared__ __bf16 Wt[2][128 * 40];              // 2 x (128 rows x 32 k, stride 40)
  const int lane = threadIdx.x & 31;
  const int wave = threadIdx.x >> 5;
  const int n0 = blockIdx.x * 128;
  const int m0 = blockIdx.y * 64 + wave * 16;
  const int r16 = lane & 15;
  v8f acc[8] = {};
  const __bf16* Arow = A + (size_t)(m0 + r16) * K;
  const __bf16* Wrow = W + (size_t)(n0 + threadIdx.x) * K;  // one W row per thread
  const unsigned lds0 = (unsigned)(size_t)(&Wt[0][threadIdx.x * 40]);
  const unsigned lds1 = (unsigned)(size_t)(&Wt[1][threadIdx.x * 40]);

  stage_async64(lds0, Wrow);                      // prologue: slab 0
  const int nIter = K >> 5;
  for (int i = 0; i < nIter; ++i) {
    const int kk = i << 5;
    wait_async0();                                // own async writes landed
    __syncthreads();                              // whole slab visible
    if (i + 1 < nIter)                            // stream next slab into other buf
      stage_async64((i & 1) ? lds0 : lds1, Wrow + kk + 32);
    if (kk + 64 < K) __builtin_prefetch(Arow + kk + 64, 0, 3);
    v16bf a = load_a_frag(Arow, kk, lane);
    const __bf16* buf = Wt[i & 1];
    v16bf b[8];
#pragma unroll
    for (int nt = 0; nt < 8; ++nt)
      b[nt] = load_b_frag(&buf[(nt * 16 + r16) * 40], 0, lane);
#pragma unroll
    for (int nt = 0; nt < 8; ++nt)
      acc[nt] = WMMA_BF16(a, b[nt], acc[nt]);
  }
  const int rhi = (lane >> 4) * 8;
#pragma unroll
  for (int nt = 0; nt < 8; ++nt)
#pragma unroll
    for (int r2 = 0; r2 < 8; ++r2)
      C[(size_t)(m0 + r2 + rhi) * N + (n0 + nt * 16 + r16)] = acc[nt][r2];
}

// ---------------------------------------------------------------------------
// RoPE + temporal cumsums.  qb/pqb: [t][h][s][128] bf16 (Pq exclusive cumsum)
// kb/skb: [t][kv][s][128] bf16, scaled by 1/sqrt(128) (Sk inclusive cumsum)
// ---------------------------------------------------------------------------
__global__ void rope_q_kernel(const float* __restrict__ qf, const int* __restrict__ pos,
                              __bf16* __restrict__ qb, __bf16* __restrict__ pqb) {
  int idx = blockIdx.x * blockDim.x + threadIdx.x;      // 16*1024*64
  if (idx >= 16 * 1024 * 64) return;
  int d2 = idx & 63, s = (idx >> 6) & 1023, h = idx >> 16;
  float ang = (float)pos[s] * __powf(10000.0f, -(float)d2 / 64.0f);
  float c = __cosf(ang), sn = __sinf(ang);
  float p1 = 0.f, p2 = 0.f;
#pragma unroll
  for (int t = 0; t < 4; ++t) {
    size_t ib = ((size_t)(t * 1024 + s)) * 2048 + h * 128;
    float x1 = qf[ib + d2], x2 = qf[ib + d2 + 64];
    float r1 = x1 * c - x2 * sn;
    float r2 = x2 * c + x1 * sn;
    size_t ob = ((size_t)((t * 16 + h) * 1024 + s)) * 128;
    qb[ob + d2] = f2bf(r1);  qb[ob + d2 + 64] = f2bf(r2);
    pqb[ob + d2] = f2bf(p1); pqb[ob + d2 + 64] = f2bf(p2);
    p1 += r1; p2 += r2;
  }
}

__global__ void rope_k_kernel(const float* __restrict__ kf, const int* __restrict__ pos,
                              __bf16* __restrict__ kb, __bf16* __restrict__ skb) {
  int idx = blockIdx.x * blockDim.x + threadIdx.x;      // 4*1024*64
  if (idx >= 4 * 1024 * 64) return;
  int d2 = idx & 63, s = (idx >> 6) & 1023, kv = idx >> 16;
  float ang = (float)pos[s] * __powf(10000.0f, -(float)d2 / 64.0f);
  float c = __cosf(ang), sn = __sinf(ang);
  const float scale = 0.08838834764831845f;             // 1/sqrt(128)
  float a1 = 0.f, a2 = 0.f;
#pragma unroll
  for (int t = 0; t < 4; ++t) {
    size_t ib = ((size_t)(t * 1024 + s)) * 512 + kv * 128;
    float x1 = kf[ib + d2], x2 = kf[ib + d2 + 64];
    float r1 = (x1 * c - x2 * sn) * scale;
    float r2 = (x2 * c + x1 * sn) * scale;
    a1 += r1; a2 += r2;
    size_t ob = ((size_t)((t * 4 + kv) * 1024 + s)) * 128;
    kb[ob + d2] = f2bf(r1);  kb[ob + d2 + 64] = f2bf(r2);
    skb[ob + d2] = f2bf(a1); skb[ob + d2 + 64] = f2bf(a2);
  }
}

// v transposed: vtb[t][kv][d][s] bf16 so attn B-frags are contiguous loads
__global__ void v_transpose_kernel(const float* __restrict__ vf, __bf16* __restrict__ vtb) {
  int idx = blockIdx.x * blockDim.x + threadIdx.x;      // 4*1024*128
  if (idx >= 4 * 1024 * 128) return;
  int d = idx & 127, s = (idx >> 7) & 1023, kv = idx >> 17;
#pragma unroll
  for (int t = 0; t < 4; ++t)
    vtb[((size_t)((t * 4 + kv) * 128 + d)) * 1024 + s] =
        f2bf(vf[((size_t)(t * 1024 + s)) * 512 + kv * 128 + d]);
}

// ---------------------------------------------------------------------------
// Attention core.  Grid: (16 m-blocks of 64 rows, 16 heads); 128 thr = 4 waves.
// Each wave owns 16 rows of X (fp32, LDS, stride 1040 for bank spread).
// Per t: Oneg = softmax(X_{t-1}) @ v_t ; X += Pq@k^T + q@Sk^T (masked);
//        stats; Opos = softmax(X_t) @ v_t ; aout_t = Opos - Oneg.
// 266KB LDS -> 1 WG per WGP -> 1 wave/SIMD, so heavy VGPR use is free.
// ---------------------------------------------------------------------------
__global__ void __launch_bounds__(128)
snn_attn_kernel(const __bf16* __restrict__ qb, const __bf16* __restrict__ pqb,
                const __bf16* __restrict__ kb, const __bf16* __restrict__ skb,
                const __bf16* __restrict__ vtb, __bf16* __restrict__ aout) {
  extern __shared__ float lds[];
  constexpr int XS = 1040;
  const int lane = threadIdx.x & 31;
  const int wave = threadIdx.x >> 5;
  const int h = blockIdx.y;
  const int hkv = h >> 2;                        // GQA: 4 heads per kv head
  const int m0 = blockIdx.x * 64 + wave * 16;
  const int r16 = lane & 15;
  const int rhi = (lane >> 4) * 8;
  float* Xw = lds + wave * 16 * XS;
  float* rowMax = lds + 4 * 16 * XS + wave * 16;
  float* rowInv = lds + 4 * 16 * XS + 64 + wave * 16;

  for (int r = 0; r < 16; ++r)
    for (int c = lane; c < 1024; c += 32) Xw[r * XS + c] = 0.f;

  for (int t = 0; t < 4; ++t) {
    v8f Oneg[8] = {};
    v8f Opos[8] = {};
    const __bf16* vtBase = vtb + (size_t)(t * 4 + hkv) * 128 * 1024;

    if (t > 0) {                                  // Y_{t-1} @ v_t from stored stats
      const float mx = rowMax[r16], inv = rowInv[r16];
      const float* Xr = Xw + r16 * XS;
      for (int kk = 0; kk < 1024; kk += 32) {
        v16bf a = y_frag(Xr, kk, lane, mx, inv);
        v16bf b[8];
#pragma unroll
        for (int dt = 0; dt < 8; ++dt)
          b[dt] = load_b_frag(vtBase + (size_t)(dt * 16 + r16) * 1024, kk, lane);
#pragma unroll
        for (int dt = 0; dt < 8; ++dt)
          Oneg[dt] = WMMA_BF16(a, b[dt], Oneg[dt]);
      }
    }

    // scores_t = Pq_t @ k_t^T + q_t @ Sk_t^T  (k pre-scaled by 1/sqrt(D))
    // A-fragments are invariant across the 64 column tiles: hoist to registers.
    const __bf16* qrow = qb  + ((size_t)(t * 16 + h) * 1024 + m0 + r16) * 128;
    const __bf16* prow = pqb + ((size_t)(t * 16 + h) * 1024 + m0 + r16) * 128;
    v16bf aP[4], aQ[4];
#pragma unroll
    for (int kq = 0; kq < 4; ++kq) {
      aP[kq] = load_a_frag(prow, kq * 32, lane);
      aQ[kq] = load_a_frag(qrow, kq * 32, lane);
    }
    for (int c = 0; c < 64; ++c) {
      v8f acc = {};
      const __bf16* krow = kb  + ((size_t)(t * 4 + hkv) * 1024 + c * 16 + r16) * 128;
      const __bf16* srow = skb + ((size_t)(t * 4 + hkv) * 1024 + c * 16 + r16) * 128;
      v16bf bK[4], bS[4];
#pragma unroll
      for (int kq = 0; kq < 4; ++kq) {
        bK[kq] = load_b_frag(krow, kq * 32, lane);
        bS[kq] = load_b_frag(srow, kq * 32, lane);
      }
#pragma unroll
      for (int kq = 0; kq < 4; ++kq) {
        acc = WMMA_BF16(aP[kq], bK[kq], acc);
        acc = WMMA_BF16(aQ[kq], bS[kq], acc);
      }
#pragma unroll
      for (int r2 = 0; r2 < 8; ++r2) {            // mask + accumulate into X
        const int wr = r2 + rhi;
        const int colg = c * 16 + r16;
        float v = acc[r2];
        if (colg > m0 + wr) v = -1.0e9f;
        Xw[wr * XS + colg] += v;
      }
    }

    // row-wise softmax stats of X_t (wave-private rows; no block barrier needed)
    for (int r = 0; r < 16; ++r) {
      const float* Xr = Xw + r * XS;
      float mx = -3.0e38f;
      for (int c2 = lane; c2 < 1024; c2 += 32) mx = fmaxf(mx, Xr[c2]);
      mx = wred_max(mx);
      float sm = 0.f;
      for (int c2 = lane; c2 < 1024; c2 += 32) sm += __expf(Xr[c2] - mx);
      sm = wred_sum(sm);
      if (lane == 0) { rowMax[r] = mx; rowInv[r] = 1.0f / sm; }
    }

    {                                             // Y_t @ v_t
      const float mx = rowMax[r16], inv = rowInv[r16];
      const float* Xr = Xw + r16 * XS;
      for (int kk = 0; kk < 1024; kk += 32) {
        v16bf a = y_frag(Xr, kk, lane, mx, inv);
        v16bf b[8];
#pragma unroll
        for (int dt = 0; dt < 8; ++dt)
          b[dt] = load_b_frag(vtBase + (size_t)(dt * 16 + r16) * 1024, kk, lane);
#pragma unroll
        for (int dt = 0; dt < 8; ++dt)
          Opos[dt] = WMMA_BF16(a, b[dt], Opos[dt]);
      }
    }

#pragma unroll
    for (int dt = 0; dt < 8; ++dt)
#pragma unroll
      for (int r2 = 0; r2 < 8; ++r2) {
        const int s = m0 + r2 + rhi;
        const int d = dt * 16 + r16;
        aout[((size_t)t * 1024 + s) * 2048 + h * 128 + d] =
            f2bf(Opos[dt][r2] - Oneg[dt][r2]);
      }
  }
}

// ---------------------------------------------------------------------------
// Host orchestration
// ---------------------------------------------------------------------------
extern "C" void kernel_launch(void* const* d_in, const int* in_sizes, int n_in,
                              void* d_out, int out_size, void* d_ws, size_t ws_size,
                              hipStream_t stream) {
  (void)in_sizes; (void)n_in; (void)out_size; (void)ws_size;
  const float* hidden = (const float*)d_in[0];
  const int*   pos    = (const int*)d_in[1];
  const float* Wq = (const float*)d_in[2];
  const float* Wk = (const float*)d_in[3];
  const float* Wv = (const float*)d_in[4];
  const float* Wo = (const float*)d_in[5];
  float* out = (float*)d_out;
  char* ws = (char*)d_ws;

  // workspace layout (bytes)
  constexpr size_t off_hb  = 0;            // 4096x2048 bf16          16 MiB
  constexpr size_t off_wqb = 16777216;     // 2048x2048 bf16           8 MiB
  constexpr size_t off_wkb = 25165824;     //  512x2048 bf16           2 MiB
  constexpr size_t off_wvb = 27262976;     //  512x2048 bf16           2 MiB
  constexpr size_t off_wob = 29360128;     // 2048x2048 bf16           8 MiB
  constexpr size_t off_qf  = 37748736;     // 4096x2048 f32           32 MiB (reused as aout)
  constexpr size_t off_kf  = 71303168;     // 4096x512  f32            8 MiB
  constexpr size_t off_vf  = 79691776;     // 4096x512  f32            8 MiB
  constexpr size_t off_qb  = 88080384;     // [4][16][1024][128] bf16 16 MiB
  constexpr size_t off_pqb = 104857600;    // same                    16 MiB
  constexpr size_t off_kb  = 121634816;    // [4][4][1024][128] bf16   4 MiB
  constexpr size_t off_skb = 125829120;    // same                     4 MiB
  constexpr size_t off_vtb = 130023424;    // [4][4][128][1024] bf16   4 MiB

  __bf16* hb   = (__bf16*)(ws + off_hb);
  __bf16* wqb  = (__bf16*)(ws + off_wqb);
  __bf16* wkb  = (__bf16*)(ws + off_wkb);
  __bf16* wvb  = (__bf16*)(ws + off_wvb);
  __bf16* wob  = (__bf16*)(ws + off_wob);
  float*  qf   = (float*)(ws + off_qf);
  float*  kf   = (float*)(ws + off_kf);
  float*  vf   = (float*)(ws + off_vf);
  __bf16* qb   = (__bf16*)(ws + off_qb);
  __bf16* pqb  = (__bf16*)(ws + off_pqb);
  __bf16* kb   = (__bf16*)(ws + off_kb);
  __bf16* skb  = (__bf16*)(ws + off_skb);
  __bf16* vtb  = (__bf16*)(ws + off_vtb);
  __bf16* aout = (__bf16*)(ws + off_qf);   // qf dead after RoPE -> reuse

  // 1) fp32 -> bf16 operands
  cvt_bf16_kernel<<<(8388608 + 255) / 256, 256, 0, stream>>>(hidden, hb, 8388608);
  cvt_bf16_kernel<<<(4194304 + 255) / 256, 256, 0, stream>>>(Wq, wqb, 4194304);
  cvt_bf16_kernel<<<(1048576 + 255) / 256, 256, 0, stream>>>(Wk, wkb, 1048576);
  cvt_bf16_kernel<<<(1048576 + 255) / 256, 256, 0, stream>>>(Wv, wvb, 1048576);
  cvt_bf16_kernel<<<(4194304 + 255) / 256, 256, 0, stream>>>(Wo, wob, 4194304);

  // 2) QKV projections (x @ W^T), fp32 out
  gemm_xwt_kernel<<<dim3(16, 64), 128, 0, stream>>>(hb, wqb, qf, 4096, 2048, 2048);
  gemm_xwt_kernel<<<dim3(4, 64), 128, 0, stream>>>(hb, wkb, kf, 4096, 512, 2048);
  gemm_xwt_kernel<<<dim3(4, 64), 128, 0, stream>>>(hb, wvb, vf, 4096, 512, 2048);

  // 3) RoPE + cumsums + v transpose, packed bf16
  rope_q_kernel<<<(1048576 + 255) / 256, 256, 0, stream>>>(qf, pos, qb, pqb);
  rope_k_kernel<<<(262144 + 255) / 256, 256, 0, stream>>>(kf, pos, kb, skb);
  v_transpose_kernel<<<(524288 + 255) / 256, 256, 0, stream>>>(vf, vtb);

  // 4) spiking attention (dynamic LDS: 4 waves * 16 rows * 1040 f32 + stats)
  const size_t attn_lds = (size_t)(4 * 16 * 1040 + 128) * sizeof(float);  // 266752 B
  snn_attn_kernel<<<dim3(16, 16), 128, attn_lds, stream>>>(qb, pqb, kb, skb, vtb, aout);

  // 5) output projection -> d_out fp32
  gemm_xwt_kernel<<<dim3(16, 64), 128, 0, stream>>>(aout, wob, out, 4096, 2048, 2048);
}
